// DepthConv_78683800862734
// MI455X (gfx1250) — compile-verified, matrix operand order checked by reading, not run on hardware
//
#include <hip/hip_runtime.h>
#include <hip/hip_bf16.h>

typedef __attribute__((ext_vector_type(16))) __bf16 v16bf;
typedef __attribute__((ext_vector_type(8)))  __bf16 v8bf;
typedef __attribute__((ext_vector_type(8)))  float  v8f;

#define ALPHA  8.3f
#define CIN    64
#define COUT   64
#define HH     256
#define WW     256
#define KK     9               // 3x3
#define KDIM   (CIN * KK)      // 576
#define KTILES (KDIM / 32)     // 18
#define NPIX   32              // pixels per block (w-chunk)
#define PSTR   592             // padded bf16 stride per pixel (16B aligned)
#define XCH    32              // x channels staged to LDS per pass
#define XCOLS  40              // padded row width (34 used: w0-1 .. w0+32)
#define XSLOTS (XCH * 3 * XCOLS)   // 3840 f32 slots per pass

// Wave-relative LDS byte offset = low 32 bits of the generic address
// (generic LDS aperture keeps the offset in addr[31:0]).
__device__ static inline unsigned lds_off(const void* p) {
    return (unsigned)(size_t)p;
}

// ---------------------------------------------------------------------------
// Pre-swizzle weights [Cout,Cin,3,3] f32 -> per-lane A-fragment layout, bf16.
// wsw[((mt*KTILES + kt)*32 + lane)*16 + h]; ISA 16-bit A 16x32 layout.
// ---------------------------------------------------------------------------
__global__ void depthconv_repack_w(const float* __restrict__ w,
                                   __bf16* __restrict__ wsw) {
    int idx = blockIdx.x * blockDim.x + threadIdx.x;
    if (idx >= 4 * KTILES * 32 * 16) return;
    int h    = idx & 15;
    int lane = (idx >> 4) & 31;
    int rest = idx >> 9;
    int kt   = rest % KTILES;
    int mt   = rest / KTILES;
    int m      = mt * 16 + (lane & 15);
    int klocal = ((h & 8) ? 16 : 0) + ((lane >= 16) ? 8 : 0) + (h & 7);
    int k      = kt * 32 + klocal;
    int c = k / KK, j = k % KK;
    wsw[idx] = (__bf16)w[(m * CIN + c) * KK + j];
}

// ---------------------------------------------------------------------------
// Main kernel: one block (256 thr = 8 waves) per (b, h, 32-wide w chunk).
// ---------------------------------------------------------------------------
__global__ __launch_bounds__(256)
void depthconv_wmma(const float* __restrict__ x,
                    const float* __restrict__ depth,
                    const __bf16* __restrict__ wsw,
                    const float* __restrict__ bias,
                    float* __restrict__ out) {
    __shared__ float  xs[XSLOTS];          // 15,360 B raw x tile (one 32-ch pass)
    __shared__ __bf16 patch[NPIX * PSTR];  // 37,888 B modulated bf16 patches
    __shared__ float  fdl[NPIX * KK];      //  1,152 B depth-similarity weights

    const int blk  = blockIdx.x;           // b*2048 + h*8 + wchunk
    const int wc   = blk & 7;
    const int hrow = (blk >> 3) & (HH - 1);
    const int b    = blk >> 11;
    const int w0   = wc * NPIX;
    const int t    = threadIdx.x;
    const int p    = t & 31;               // pixel owned in phase 2
    const int g    = t >> 5;               // channel group (8 groups of 4)

    const float* xb = x + (size_t)b * CIN * HH * WW;

    // ---- async-stage x channels [s*32, s*32+32) into LDS (ASYNCcnt path) ----
    auto stage = [&](int s) {
        for (int idx = t; idx < XSLOTS; idx += 256) {
            int col = idx % XCOLS;
            if (col >= 34) continue;                 // pad cols never read
            int row = idx / XCOLS;
            int cl  = row / 3, r = row % 3;
            int hn  = hrow + r - 1;
            int wn  = w0 + col - 1;
            if (hn >= 0 && hn < HH && wn >= 0 && wn < WW) {
                unsigned long long ga = (unsigned long long)(size_t)
                    (xb + ((size_t)(s * XCH + cl) * HH + hn) * WW + wn);
                unsigned la = lds_off(&xs[idx]);
                asm volatile("global_load_async_to_lds_b32 %0, %1, off"
                             :: "v"(la), "v"(ga) : "memory");
            } else {
                xs[idx] = 0.0f;                      // zero padding (reference)
            }
        }
    };

    // pass 0 async loads in flight while we compute fd
    stage(0);

    // ---- fd[pix][j] = exp(-alpha*|d_neighbor - d_center|) ----
    const float* drow = depth + (size_t)b * HH * WW;
    for (int idx = t; idx < NPIX * KK; idx += 256) {
        int pp = idx / KK, j = idx % KK;
        int dy = j / 3 - 1, dx = j % 3 - 1;
        int hw = w0 + pp;
        float center = drow[hrow * WW + hw];
        int hn = hrow + dy, wn = hw + dx;
        float dn = (hn >= 0 && hn < HH && wn >= 0 && wn < WW)
                       ? drow[hn * WW + wn] : 0.0f;
        fdl[idx] = __expf(-ALPHA * fabsf(dn - center));
    }

    asm volatile("s_wait_asynccnt 0" ::: "memory");
    __syncthreads();

    // fd for this thread's pixel -> registers (reused across all 64 channels)
    float fdr[KK];
    #pragma unroll
    for (int j = 0; j < KK; ++j) fdr[j] = fdl[p * KK + j];

    // ---- build modulated bf16 patches in two 32-channel passes ----
    #pragma unroll
    for (int s = 0; s < 2; ++s) {
        if (s == 1) {                  // xs consumed by all waves; restage
            __syncthreads();
            stage(1);
            asm volatile("s_wait_asynccnt 0" ::: "memory");
            __syncthreads();
        }
        #pragma unroll
        for (int ci = 0; ci < 4; ++ci) {
            int cl = g * 4 + ci;                       // staged channel 0..31
            int c  = s * XCH + cl;                     // global channel
            const float* xrow = &xs[cl * 3 * XCOLS + p + 1];
            __bf16* pp = &patch[p * PSTR + c * KK];
            #pragma unroll
            for (int j = 0; j < KK; ++j) {
                int dy = j / 3, dx = (j % 3) - 1;      // compile-time offsets
                float xv = xrow[dy * XCOLS + dx];
                pp[j] = (__bf16)(xv * fdr[j]);
            }
        }
    }
    __syncthreads();

    // ---- GEMM: 64x32 tile, K=576, v_wmma_f32_16x16x32_bf16 ----
    const int lane = t & 31;
    const int wave = t >> 5;
    const int mt   = wave >> 1;              // 0..3 : Cout tile
    const int nt   = wave & 1;               // 0..1 : pixel tile
    const int n    = nt * 16 + (lane & 15);  // pixel column owned by this lane
    const int kb   = (lane >= 16) ? 8 : 0;   // ISA B-layout K base per half-wave

    v8f acc = {};
    const v16bf* afrags = (const v16bf*)wsw + (size_t)mt * KTILES * 32 + lane;
    #pragma unroll
    for (int kt = 0; kt < KTILES; ++kt) {
        v16bf a = afrags[kt * 32];           // 32B contiguous per lane (L2-hot)
        const __bf16* bp = &patch[n * PSTR + kt * 32 + kb];
        v8bf blo = *(const v8bf*)bp;         // ds_load_b128: K kb..kb+7
        v8bf bhi = *(const v8bf*)(bp + 16);  // ds_load_b128: K kb+16..kb+23
        v16bf bm = __builtin_shufflevector(blo, bhi,
                     0, 1, 2, 3, 4, 5, 6, 7, 8, 9, 10, 11, 12, 13, 14, 15);
        acc = __builtin_amdgcn_wmma_f32_16x16x32_bf16(
                  false, a, false, bm, (short)0, acc, false, false);
    }

    // ---- epilogue: C layout -> out[b][o][h][w] + bias ----
    const int obase = mt * 16 + ((lane < 16) ? 0 : 8);
    const float4* b4 = (const float4*)(bias + obase);   // obase % 8 == 0
    float4 bl = b4[0], bh = b4[1];
    float bv[8] = { bl.x, bl.y, bl.z, bl.w, bh.x, bh.y, bh.z, bh.w };
    const size_t outb = (size_t)b * COUT * HH * WW;
    #pragma unroll
    for (int r = 0; r < 8; ++r) {
        int o = obase + r;
        out[outb + ((size_t)o * HH + hrow) * WW + (w0 + n)] = acc[r] + bv[r];
    }
}

extern "C" void kernel_launch(void* const* d_in, const int* in_sizes, int n_in,
                              void* d_out, int out_size, void* d_ws, size_t ws_size,
                              hipStream_t stream) {
    const float* x      = (const float*)d_in[0];  // [8,64,256,256]
    const float* depth  = (const float*)d_in[1];  // [8,1,256,256]
    const float* weight = (const float*)d_in[2];  // [64,64,3,3]
    const float* bias   = (const float*)d_in[3];  // [64]
    float*       out    = (float*)d_out;          // [8,64,256,256]
    __bf16*      wsw    = (__bf16*)d_ws;          // 73,728 B swizzled weights

    const int nrep = 4 * KTILES * 32 * 16;
    depthconv_repack_w<<<(nrep + 255) / 256, 256, 0, stream>>>(weight, wsw);

    const int nblk = 8 * HH * (WW / NPIX);        // 16384
    depthconv_wmma<<<nblk, 256, 0, stream>>>(x, depth, wsw, bias, out);
}
